// ASIS_20675972563782
// MI455X (gfx1250) — matrix operand order repaired and checked by reading, not compiled, not run on hardware
//
#include <hip/hip_runtime.h>
#include <hip/hip_bf16.h>
#include <stdint.h>

// ---------------------------------------------------------------------------
// Problem constants (must match reference)
// ---------------------------------------------------------------------------
#define B_      8
#define N_      4096
#define KNN     30
#define H_      128
#define CIN     256
#define SEMOUT  13
#define INSOUT  5
#define BN_EPS  1e-5f

typedef __attribute__((ext_vector_type(16))) __bf16 v16bf;
typedef __attribute__((ext_vector_type(8)))  float  v8f;

// ---------------------------------------------------------------------------
// Workspace layout (byte offsets into d_ws)
// ---------------------------------------------------------------------------
#define OFF_WSEM   0u            // 128*256 bf16 = 65536 B   (BN-folded W_sem)
#define OFF_WAD    65536u        // 128*128 bf16 = 32768 B
#define OFF_WINS   98304u        // 128*256 bf16 = 65536 B
#define OFF_BSEM   163840u       // 128 f32 folded biases
#define OFF_BAD    164352u
#define OFF_BINS   164864u
#define OFF_XSEMT  1048576u      // (B,N,256) bf16 = 16 MB   (x_sem transposed)
#define OFF_XINST  17825792u     // (B,N,256) bf16 = 16 MB   (x_ins transposed)
#define OFF_FSEMT  34603008u     // (B,N,128) bf16 =  8 MB   (relu f_sem, transposed)
#define OFF_FSINS  42991616u     // (B,128,N) f32  = 16 MB   (f_ins + adapted)
#define OFF_SQ     59768832u     // (B,N) f32 = 128 KB
#define OFF_IDX    59899904u     // (B,N,KNN) int = 3.93 MB
#define OFF_FISEM  63832064u     // (B,128,N) f32 = 16 MB

// ---------------------------------------------------------------------------
// Fold BN (eval) into conv weights/bias and convert weights to bf16.
//   y = relu((s*W) x + (s*(b-mu)+beta)),  s = g/sqrt(var+eps)
// ---------------------------------------------------------------------------
__global__ void asis_prep_weights(
    const float* __restrict__ Wsem, const float* __restrict__ bsem,
    const float* __restrict__ gsem, const float* __restrict__ besem,
    const float* __restrict__ musem, const float* __restrict__ vasem,
    const float* __restrict__ Wad,  const float* __restrict__ bad_,
    const float* __restrict__ gad,  const float* __restrict__ bead,
    const float* __restrict__ muad, const float* __restrict__ vaad,
    const float* __restrict__ Wins, const float* __restrict__ bins,
    const float* __restrict__ gins, const float* __restrict__ beins,
    const float* __restrict__ muins,const float* __restrict__ vains,
    __bf16* __restrict__ WsemB, __bf16* __restrict__ WadB, __bf16* __restrict__ WinsB,
    float* __restrict__ bsemF, float* __restrict__ badF, float* __restrict__ binsF)
{
    int o = threadIdx.x;
    if (o >= H_) return;
    float s0 = gsem[o] * rsqrtf(vasem[o] + BN_EPS);
    bsemF[o] = (bsem[o] - musem[o]) * s0 + besem[o];
    for (int i = 0; i < CIN; ++i) WsemB[o * CIN + i] = (__bf16)(Wsem[o * CIN + i] * s0);
    float s1 = gad[o] * rsqrtf(vaad[o] + BN_EPS);
    badF[o] = (bad_[o] - muad[o]) * s1 + bead[o];
    for (int i = 0; i < H_; ++i)  WadB[o * H_ + i]  = (__bf16)(Wad[o * H_ + i] * s1);
    float s2 = gins[o] * rsqrtf(vains[o] + BN_EPS);
    binsF[o] = (bins[o] - muins[o]) * s2 + beins[o];
    for (int i = 0; i < CIN; ++i) WinsB[o * CIN + i] = (__bf16)(Wins[o * CIN + i] * s2);
}

// ---------------------------------------------------------------------------
// Transpose + convert: (B, C=256, N) f32  ->  (B, N, C) bf16, packed stores.
// One thread per (b, n).
// ---------------------------------------------------------------------------
__global__ void asis_cvt_t256(const float* __restrict__ X, __bf16* __restrict__ XT)
{
    int t = blockIdx.x * blockDim.x + threadIdx.x;      // B_*N_ threads exactly
    int b = t >> 12;
    int n = t & 4095;
    const float* col = X + (size_t)b * CIN * N_ + n;
    uint32_t* dst = (uint32_t*)(XT + (size_t)t * CIN);
    for (int c = 0; c < CIN; c += 8) {
        union { uint32_t u[4]; __bf16 h[8]; } pk;
#pragma unroll
        for (int e = 0; e < 8; ++e) pk.h[e] = (__bf16)col[(size_t)(c + e) * N_];
        dst[(c >> 1) + 0] = pk.u[0];
        dst[(c >> 1) + 1] = pk.u[1];
        dst[(c >> 1) + 2] = pk.u[2];
        dst[(c >> 1) + 3] = pk.u[3];
    }
}

// ---------------------------------------------------------------------------
// WMMA fragment loaders (wave32; contiguous k order k = k0 + 16*(L/16) + e, same in A and B)
// ---------------------------------------------------------------------------
// A (16x32 bf16): lane L holds row M; a[0..15]=W[M][k0+kh+0..15],
// kh=(L>=16)?16:0.  Two 16B chunks.
__device__ __forceinline__ v16bf asis_load_afrag(const __bf16* wrow, int k0, int khalf)
{
    union { v16bf v; uint32_t u[8]; } f;
    const uint32_t* p0 = (const uint32_t*)(wrow + k0 + 2 * khalf);
    const uint32_t* p1 = (const uint32_t*)(wrow + k0 + 2 * khalf + 8);
    f.u[0] = p0[0]; f.u[1] = p0[1]; f.u[2] = p0[2]; f.u[3] = p0[3];
    f.u[4] = p1[0]; f.u[5] = p1[1]; f.u[6] = p1[2]; f.u[7] = p1[3];
    return f.v;
}

// B (32x16 bf16) from K-contiguous transposed activations:
// lane L holds column N; element e holds K = k0 + e + ((L>=16)?16:0)
// -> 16 consecutive bf16 = one 32B contiguous block.
__device__ __forceinline__ v16bf asis_load_bfrag(const __bf16* p)
{
    union { v16bf v; uint32_t u[8]; } f;
    const uint32_t* q = (const uint32_t*)p;
#pragma unroll
    for (int e = 0; e < 8; ++e) f.u[e] = q[e];
    return f.v;
}

// ---------------------------------------------------------------------------
// GEMM 1: relu(BN(W_sem @ x_sem)) -> FSEMT (B,N,128) bf16 (transposed).
// One wave: all 8 M-tiles of one 16-column strip (8 accumulators);
// B fragment loaded once per K-step and reused by 8 WMMAs.
// ---------------------------------------------------------------------------
__global__ __launch_bounds__(128) void asis_gemm_sem(
    const __bf16* __restrict__ XT, const __bf16* __restrict__ Wb,
    const float* __restrict__ bias, __bf16* __restrict__ outT)
{
    int lane = threadIdx.x & 31;
    int wave = threadIdx.x >> 5;
    int tile = blockIdx.x * 4 + wave;       // B_*(N_/16) = 2048 wave-tiles
    int b  = tile >> 8;
    int Nbase = (tile & 255) << 4;
    int nIdx  = Nbase + (lane & 15);
    int khalf = (lane >= 16) ? 8  : 0;
    int kboff = (lane >= 16) ? 16 : 0;
    int mhi   = (lane >= 16) ? 8  : 0;

    const __bf16* xrow = XT + ((size_t)b * N_ + nIdx) * CIN;   // K-contiguous
    const __bf16* wl   = Wb + (size_t)(lane & 15) * CIN;       // row (lane&15)

    v8f acc[8];
#pragma unroll
    for (int mt = 0; mt < 8; ++mt) acc[mt] = (v8f){};

#pragma unroll
    for (int k0 = 0; k0 < CIN; k0 += 32) {
        v16bf bm = asis_load_bfrag(xrow + k0 + kboff);
#pragma unroll
        for (int mt = 0; mt < 8; ++mt) {
            v16bf a = asis_load_afrag(wl + (size_t)mt * 16 * CIN, k0, khalf);
            acc[mt] = __builtin_amdgcn_wmma_f32_16x16x32_bf16(
                false, a, false, bm, (short)0, acc[mt], false, false);
        }
    }

    __bf16* orow = outT + ((size_t)b * N_ + nIdx) * H_;
#pragma unroll
    for (int mt = 0; mt < 8; ++mt) {
        union { uint32_t u[4]; __bf16 h[8]; } pk;
#pragma unroll
        for (int rr = 0; rr < 8; ++rr) {
            int M = mt * 16 + rr + mhi;
            float v = acc[mt][rr] + bias[M];
            pk.h[rr] = (__bf16)(v > 0.f ? v : 0.f);
        }
        uint32_t* q = (uint32_t*)(orow + mt * 16 + mhi);
        q[0] = pk.u[0]; q[1] = pk.u[1]; q[2] = pk.u[2]; q[3] = pk.u[3];
    }
}

// ---------------------------------------------------------------------------
// GEMM 2 (fused): f_sins = relu(BN(W_ins @ x_ins)) + relu(BN(W_ad @ f_sem))
// Phase A: K=256 over XINST; phase B: K=128 over FSEMT. Output f32 M-major.
// ---------------------------------------------------------------------------
__global__ __launch_bounds__(128) void asis_gemm_fused2(
    const __bf16* __restrict__ XinsT, const __bf16* __restrict__ FsemT,
    const __bf16* __restrict__ WinsB, const __bf16* __restrict__ WadB,
    const float* __restrict__ binsF, const float* __restrict__ badF,
    float* __restrict__ fsins)
{
    int lane = threadIdx.x & 31;
    int wave = threadIdx.x >> 5;
    int tile = blockIdx.x * 4 + wave;
    int b  = tile >> 8;
    int Nbase = (tile & 255) << 4;
    int nIdx  = Nbase + (lane & 15);
    int khalf = (lane >= 16) ? 8  : 0;
    int kboff = (lane >= 16) ? 16 : 0;
    int mhi   = (lane >= 16) ? 8  : 0;

    // Phase A: W_ins (128x256) @ x_ins
    const __bf16* xrow2 = XinsT + ((size_t)b * N_ + nIdx) * CIN;
    const __bf16* wl2   = WinsB + (size_t)(lane & 15) * CIN;
    v8f acc[8];
#pragma unroll
    for (int mt = 0; mt < 8; ++mt) acc[mt] = (v8f){};
#pragma unroll
    for (int k0 = 0; k0 < CIN; k0 += 32) {
        v16bf bm = asis_load_bfrag(xrow2 + k0 + kboff);
#pragma unroll
        for (int mt = 0; mt < 8; ++mt) {
            v16bf a = asis_load_afrag(wl2 + (size_t)mt * 16 * CIN, k0, khalf);
            acc[mt] = __builtin_amdgcn_wmma_f32_16x16x32_bf16(
                false, a, false, bm, (short)0, acc[mt], false, false);
        }
    }
    // acc <- relu(acc + bins)
#pragma unroll
    for (int mt = 0; mt < 8; ++mt)
#pragma unroll
        for (int rr = 0; rr < 8; ++rr) {
            float v = acc[mt][rr] + binsF[mt * 16 + rr + mhi];
            acc[mt][rr] = v > 0.f ? v : 0.f;
        }

    // Phase B: W_ad (128x128) @ relu(f_sem)
    const __bf16* xrow1 = FsemT + ((size_t)b * N_ + nIdx) * H_;
    const __bf16* wl1   = WadB + (size_t)(lane & 15) * H_;
    v8f acd[8];
#pragma unroll
    for (int mt = 0; mt < 8; ++mt) acd[mt] = (v8f){};
#pragma unroll
    for (int k0 = 0; k0 < H_; k0 += 32) {
        v16bf bm = asis_load_bfrag(xrow1 + k0 + kboff);
#pragma unroll
        for (int mt = 0; mt < 8; ++mt) {
            v16bf a = asis_load_afrag(wl1 + (size_t)mt * 16 * H_, k0, khalf);
            acd[mt] = __builtin_amdgcn_wmma_f32_16x16x32_bf16(
                false, a, false, bm, (short)0, acd[mt], false, false);
        }
    }

    float* ob = fsins + (size_t)b * H_ * N_;
#pragma unroll
    for (int mt = 0; mt < 8; ++mt)
#pragma unroll
        for (int rr = 0; rr < 8; ++rr) {
            int M = mt * 16 + rr + mhi;
            float v2 = acd[mt][rr] + badF[M];
            v2 = v2 > 0.f ? v2 : 0.f;                 // relu(adapted)
            ob[(size_t)M * N_ + nIdx] = acc[mt][rr] + v2;
        }
}

// ---------------------------------------------------------------------------
// e_ins = W_emb @ f_sins + b_emb  (5x128 per column), plus |e|^2 per point.
// ---------------------------------------------------------------------------
__global__ void asis_emb_sq(
    const float* __restrict__ fsins, const float* __restrict__ Wemb,
    const float* __restrict__ bemb, float* __restrict__ eout,
    float* __restrict__ sq)
{
    int t = blockIdx.x * blockDim.x + threadIdx.x;      // B_*N_ threads exactly
    int b = t >> 12;
    int n = t & 4095;
    const float* col = fsins + (size_t)b * H_ * N_ + n;
    float acc[INSOUT];
#pragma unroll
    for (int o = 0; o < INSOUT; ++o) acc[o] = bemb[o];
    for (int c = 0; c < H_; ++c) {
        float xv = col[(size_t)c * N_];
#pragma unroll
        for (int o = 0; o < INSOUT; ++o) acc[o] += Wemb[o * H_ + c] * xv;
    }
    float s = 0.f;
#pragma unroll
    for (int o = 0; o < INSOUT; ++o) {
        eout[((size_t)b * INSOUT + o) * N_ + n] = acc[o];
        s += acc[o] * acc[o];
    }
    sq[t] = s;
}

// ---------------------------------------------------------------------------
// kNN top-30. One wave32 per query row, 4 waves/block (all same batch).
// LDS: [0, 64KB) = 4 x 4096-float distance rows; [64KB, 80KB) = batch |e|^2
// staged with CDNA5 GLOBAL_LOAD_ASYNC_TO_LDS_B128 (ASYNCcnt-fenced).
// ---------------------------------------------------------------------------
__global__ __launch_bounds__(128) void asis_knn_topk(
    const float* __restrict__ e, const float* __restrict__ sq,
    int* __restrict__ nnidx)
{
    extern __shared__ float smem[];
    int lane = threadIdx.x & 31;
    int wave = threadIdx.x >> 5;
    int row  = blockIdx.x * 4 + wave;       // B_*N_ rows; batch uniform in block
    int b = row >> 12;
    int i = row & 4095;
    float* dist = smem + (size_t)wave * N_;
    const float* sqs = smem + 4 * N_;       // staged |e|^2 for this batch

    // ---- async stage sq[b][0..N) into LDS (1024 x 16B chunks, whole block)
    {
        const float* sqb = sq + (size_t)b * N_;
        for (int it = 0; it < 8; ++it) {
            int chunk = threadIdx.x + 128 * it;               // 0..1023
            int ldsOff = 4 * N_ * 4 + chunk * 16;             // byte addr in LDS
            const float* g = sqb + chunk * 4;
            asm volatile("global_load_async_to_lds_b128 %0, %1, off"
                         :: "v"(ldsOff), "v"(g) : "memory");
        }
        asm volatile("s_wait_asynccnt 0" ::: "memory");
    }
    __syncthreads();

    const float* eb = e + (size_t)b * INSOUT * N_;
    float e0 = eb[i], e1 = eb[N_ + i], e2 = eb[2 * N_ + i],
          e3 = eb[3 * N_ + i], e4 = eb[4 * N_ + i];
    float sqi = sqs[i];

    for (int j = lane; j < N_; j += 32) {
        float d = sqi + sqs[j]
                - 2.f * (e0 * eb[j] + e1 * eb[N_ + j] + e2 * eb[2 * N_ + j]
                       + e3 * eb[3 * N_ + j] + e4 * eb[4 * N_ + j]);
        dist[j] = d;
    }
    __syncthreads();

    int* orow = nnidx + (size_t)(b * N_ + i) * KNN;
    for (int t = 0; t < KNN; ++t) {
        float bv = __builtin_inff();
        int   bi = N_;
        for (int j = lane; j < N_; j += 32) {
            float v = dist[j];
            if (v < bv) { bv = v; bi = j; }     // increasing j keeps lowest idx
        }
#pragma unroll
        for (int off = 16; off > 0; off >>= 1) {
            float ov = __shfl_xor(bv, off, 32);
            int   oi = __shfl_xor(bi, off, 32);
            if (ov < bv || (ov == bv && oi < bi)) { bv = ov; bi = oi; }
        }
        if (lane == 0) { orow[t] = bi; dist[bi] = __builtin_inff(); }
        __syncthreads();
    }
}

// ---------------------------------------------------------------------------
// f_isem[b][c][n] = max_k relu(f_sem)[b][c][ nn_idx[b][n][k] ]  (from FSEMT)
// ---------------------------------------------------------------------------
__global__ void asis_gather_max(
    const __bf16* __restrict__ fsemT, const int* __restrict__ nnidx,
    float* __restrict__ fisem)
{
    int t = blockIdx.x * blockDim.x + threadIdx.x;      // B_*H_*N_ exactly
    int n = t & 4095;
    int rest = t >> 12;
    int c = rest & 127;
    int b = rest >> 7;
    const int* irow = nnidx + (size_t)(b * N_ + n) * KNN;
    __builtin_prefetch(irow, 0, 3);
    const __bf16* f = fsemT + (size_t)b * N_ * H_ + c;
    float m = -__builtin_inff();
#pragma unroll 5
    for (int k = 0; k < KNN; ++k)
        m = fmaxf(m, (float)f[(size_t)irow[k] * H_]);
    fisem[((size_t)b * H_ + c) * N_ + n] = m;
}

// ---------------------------------------------------------------------------
// p_sem = W_pred @ f_isem + b_pred   (13x128 per column)
// ---------------------------------------------------------------------------
__global__ void asis_pred(
    const float* __restrict__ fisem, const float* __restrict__ Wp,
    const float* __restrict__ bp, float* __restrict__ out)
{
    int t = blockIdx.x * blockDim.x + threadIdx.x;      // B_*SEMOUT*N_ exactly
    int n = t & 4095;
    int rest = t >> 12;
    int o = rest % SEMOUT;
    int b = rest / SEMOUT;
    const float* col = fisem + (size_t)b * H_ * N_ + n;
    const float* w   = Wp + o * H_;
    float acc = bp[o];
    for (int c = 0; c < H_; ++c) acc += w[c] * col[(size_t)c * N_];
    out[((size_t)b * SEMOUT + o) * N_ + n] = acc;
}

// ---------------------------------------------------------------------------
extern "C" void kernel_launch(void* const* d_in, const int* in_sizes, int n_in,
                              void* d_out, int out_size, void* d_ws, size_t ws_size,
                              hipStream_t stream)
{
    (void)in_sizes; (void)n_in; (void)out_size; (void)ws_size;

    const float* f_sem  = (const float*)d_in[0];
    const float* f_ins  = (const float*)d_in[1];
    const float* W_sem  = (const float*)d_in[2];
    const float* b_sem  = (const float*)d_in[3];
    const float* g_sem  = (const float*)d_in[4];
    const float* be_sem = (const float*)d_in[5];
    const float* mu_sem = (const float*)d_in[6];
    const float* va_sem = (const float*)d_in[7];
    const float* W_ad   = (const float*)d_in[8];
    const float* b_ad   = (const float*)d_in[9];
    const float* g_ad   = (const float*)d_in[10];
    const float* be_ad  = (const float*)d_in[11];
    const float* mu_ad  = (const float*)d_in[12];
    const float* va_ad  = (const float*)d_in[13];
    const float* W_ins  = (const float*)d_in[14];
    const float* b_ins  = (const float*)d_in[15];
    const float* g_ins  = (const float*)d_in[16];
    const float* be_ins = (const float*)d_in[17];
    const float* mu_ins = (const float*)d_in[18];
    const float* va_ins = (const float*)d_in[19];
    const float* W_emb  = (const float*)d_in[20];
    const float* b_emb  = (const float*)d_in[21];
    const float* W_pred = (const float*)d_in[22];
    const float* b_pred = (const float*)d_in[23];

    char* ws = (char*)d_ws;
    __bf16* WsemB = (__bf16*)(ws + OFF_WSEM);
    __bf16* WadB  = (__bf16*)(ws + OFF_WAD);
    __bf16* WinsB = (__bf16*)(ws + OFF_WINS);
    float* bsemF  = (float*)(ws + OFF_BSEM);
    float* badF   = (float*)(ws + OFF_BAD);
    float* binsF  = (float*)(ws + OFF_BINS);
    __bf16* XsemT = (__bf16*)(ws + OFF_XSEMT);
    __bf16* XinsT = (__bf16*)(ws + OFF_XINST);
    __bf16* FsemT = (__bf16*)(ws + OFF_FSEMT);
    float* fsins  = (float*)(ws + OFF_FSINS);
    float* sqW    = (float*)(ws + OFF_SQ);
    int*   idxW   = (int*)  (ws + OFF_IDX);
    float* fisem  = (float*)(ws + OFF_FISEM);

    float* p_sem = (float*)d_out;                               // (B,13,N)
    float* e_out = p_sem + (size_t)B_ * SEMOUT * N_;            // (B,5,N)

    asis_prep_weights<<<1, 128, 0, stream>>>(
        W_sem, b_sem, g_sem, be_sem, mu_sem, va_sem,
        W_ad,  b_ad,  g_ad,  be_ad,  mu_ad,  va_ad,
        W_ins, b_ins, g_ins, be_ins, mu_ins, va_ins,
        WsemB, WadB, WinsB, bsemF, badF, binsF);

    asis_cvt_t256<<<(B_ * N_) / 256, 256, 0, stream>>>(f_sem, XsemT);
    asis_cvt_t256<<<(B_ * N_) / 256, 256, 0, stream>>>(f_ins, XinsT);

    // 2048 wave-tiles / 4 waves per block
    asis_gemm_sem<<<512, 128, 0, stream>>>(XsemT, WsemB, bsemF, FsemT);
    asis_gemm_fused2<<<512, 128, 0, stream>>>(XinsT, FsemT, WinsB, WadB,
                                              binsF, badF, fsins);

    asis_emb_sq<<<(B_ * N_) / 256, 256, 0, stream>>>(fsins, W_emb, b_emb,
                                                     e_out, sqW);

    asis_knn_topk<<<(B_ * N_) / 4, 128, (4 * N_ + N_) * sizeof(float), stream>>>(
        e_out, sqW, idxW);

    asis_gather_max<<<(B_ * H_ * N_) / 256, 256, 0, stream>>>(FsemT, idxW, fisem);

    asis_pred<<<(B_ * SEMOUT * N_) / 256, 256, 0, stream>>>(fisem, W_pred,
                                                            b_pred, p_sem);
}
